// Gemma3Atttention_33741263077402
// MI455X (gfx1250) — matrix operand order repaired and checked by reading, not compiled, not run on hardware
//
#include <hip/hip_runtime.h>

// ---------------------------------------------------------------------------
// CDNA5 (gfx1250) wave32 WMMA + TDM implementation of Gemma3 attention layer.
// B=2, L=2048, D=2048, H=8, HD=256. fp32 I/O, bf16 WMMA compute.
// ---------------------------------------------------------------------------

typedef __attribute__((ext_vector_type(16))) __bf16 v16bf;
typedef __attribute__((ext_vector_type(8)))  float  v8f;
typedef __attribute__((ext_vector_type(4)))  unsigned int u32x4;
typedef __attribute__((ext_vector_type(8)))  int    i32x8;
typedef __attribute__((ext_vector_type(4)))  int    i32x4;
typedef unsigned int  u32;
typedef unsigned short u16;
typedef unsigned long long u64;

union FragBF {
    v16bf v;
    u32   u[8];
    uint4 q4[2];
};

static __device__ inline v8f wmma_bf16(const FragBF& a, const FragBF& b, v8f c) {
    return __builtin_amdgcn_wmma_f32_16x16x32_bf16(
        /*neg_a=*/false, a.v, /*neg_b=*/false, b.v,
        /*c_mod=*/(short)0, c, /*reuse_a=*/false, /*reuse_b=*/false);
}

static __device__ inline u16 f2bf(float f) {
    u32 u = __builtin_bit_cast(u32, f);
    u32 r = (u + 0x7FFFu + ((u >> 16) & 1u)) >> 16;
    return (u16)r;
}

static __device__ inline v8f v8f_zero() {
    v8f z = {0.f, 0.f, 0.f, 0.f, 0.f, 0.f, 0.f, 0.f};
    return z;
}

// ---------------------------------------------------------------------------
// TDM: 2-D tile load (bf16 elements) Global -> LDS via Tensor Data Mover.
// tile_w elems along dim0 (contiguous), tile_h rows, row stride in elems.
// D# bitfields per CDNA5 ISA ch.8 (groups 0/1; 2D so groups 2/3 zero).
// ---------------------------------------------------------------------------
static __device__ inline void tdm_load_2d_bf16(
    unsigned lds_off, const u16* gptr,
    int tile_w, int tile_h, int row_stride,
    int tensor_w, int tensor_h)
{
    u64 ga = (u64)(uintptr_t)gptr;
    u32x4 g0;
    g0[0] = 1u;                                   // count=1, user mode, no gather
    g0[1] = lds_off;                              // lds_addr
    g0[2] = (u32)(ga & 0xFFFFFFFFu);              // global_addr[31:0]
    g0[3] = (u32)((ga >> 32) & 0x1FFFFFFu)        // global_addr[56:32]
          | (2u << 30);                           // type=2 ("image")

    i32x8 g1;
    g1[0] = (int)(1u << 16);                                  // data_size=1 (2B)
    g1[1] = (int)(((u32)tensor_w & 0xFFFFu) << 16);           // tensor_dim0 lo16
    g1[2] = (int)(((u32)tensor_w >> 16) |
                  (((u32)tensor_h & 0xFFFFu) << 16));         // dim0 hi / dim1 lo
    g1[3] = (int)(((u32)tensor_h >> 16) |
                  ((u32)tile_w << 16));                       // dim1 hi / tile_dim0
    g1[4] = (int)((u32)tile_h & 0xFFFFu);                     // tile_dim1 (tile_dim2=0)
    g1[5] = (int)(u32)row_stride;                             // tensor_dim0_stride[31:0]
    g1[6] = 0;                                                // stride hi / dim1_stride lo
    g1[7] = 0;

    i32x4 z4 = {0, 0, 0, 0};
    i32x8 z8 = {0, 0, 0, 0, 0, 0, 0, 0};
    __builtin_amdgcn_tensor_load_to_lds(g0, g1, z4, z4, z8, 0);
}

// ---------------------------------------------------------------------------
// fp32 -> bf16 elementwise convert (one-shot; removes conversions from GEMM).
// ---------------------------------------------------------------------------
__global__ __launch_bounds__(256) void cvt_bf16_kernel(
    const float* __restrict__ in, u16* __restrict__ out, int n)
{
    int i = (blockIdx.x * 256 + threadIdx.x) * 8;
    if (i + 8 <= n) {
        float4 a = *(const float4*)&in[i];
        float4 b = *(const float4*)&in[i + 4];
        u16 t[8] = { f2bf(a.x), f2bf(a.y), f2bf(a.z), f2bf(a.w),
                     f2bf(b.x), f2bf(b.y), f2bf(b.z), f2bf(b.w) };
        *(uint4*)&out[i] = *(const uint4*)t;
    }
}

// ---------------------------------------------------------------------------
// All-bf16 GEMM: C[M,N] = A[M,K] * W[N,K]^T, fp32 accumulate/output.
// Block tile 128x128, BK=32, 256 threads (8 waves), wave tile 64x32.
// Tile staging via TDM (tensor_load_to_lds), double-buffered: wave 0 issues
// stage i+1's DMA, waits tensorcnt<=2 (stage i resident; TDM ops of one wave
// complete in order), then all waves compute stage i while DMA runs.
// ---------------------------------------------------------------------------
__global__ __launch_bounds__(256) void gemm_bf16_kernel(
    const u16* __restrict__ A, const u16* __restrict__ W,
    float* __restrict__ C, int M, int N, int K)
{
    __shared__ __align__(16) u16 As[2][128 * 32];
    __shared__ __align__(16) u16 Bs[2][128 * 32];

    const int tid  = threadIdx.x;
    const int lane = tid & 31;
    const int w    = tid >> 5;
    const int half = lane >> 4;
    const int l16  = lane & 15;
    const int wm   = w & 1;        // 0..1  -> 64-row slab
    const int wn   = w >> 1;       // 0..3  -> 32-col slab
    const int m0   = blockIdx.y * 128;
    const int n0   = blockIdx.x * 128;

    v8f acc[4][2];
#pragma unroll
    for (int i = 0; i < 4; ++i)
#pragma unroll
        for (int j = 0; j < 2; ++j) acc[i][j] = v8f_zero();

    const unsigned asOff0 = (unsigned)(uintptr_t)&As[0][0];
    const unsigned asOff1 = (unsigned)(uintptr_t)&As[1][0];
    const unsigned bsOff0 = (unsigned)(uintptr_t)&Bs[0][0];
    const unsigned bsOff1 = (unsigned)(uintptr_t)&Bs[1][0];

    const int nk = K >> 5;                       // K / 32

    if (w == 0) {                                // prologue: stage 0 DMA
        tdm_load_2d_bf16(asOff0, A + (size_t)m0 * K, 32, 128, K, K, M);
        tdm_load_2d_bf16(bsOff0, W + (size_t)n0 * K, 32, 128, K, K, N);
    }

    for (int it = 0; it < nk; ++it) {
        const int cur = it & 1;
        // All waves must be done reading buffer cur^1 before DMA overwrites it.
        __syncthreads();
        if (w == 0) {
            if (it + 1 < nk) {
                int k1 = (it + 1) << 5;
                tdm_load_2d_bf16(cur ? asOff0 : asOff1,
                                 A + (size_t)m0 * K + k1, 32, 128, K, K, M);
                tdm_load_2d_bf16(cur ? bsOff0 : bsOff1,
                                 W + (size_t)n0 * K + k1, 32, 128, K, K, N);
                __builtin_amdgcn_s_wait_tensorcnt(2);   // stage it resident
            } else {
                __builtin_amdgcn_s_wait_tensorcnt(0);
            }
        }
        __syncthreads();

        // ---- B fragments (shared across the 4 A fragments) ----
        FragBF bfr[2];
#pragma unroll
        for (int j = 0; j < 2; ++j) {
            int nrow = wn * 32 + j * 16 + l16;
            int base = nrow * 32 + half * 8;           // K = half*8 + 2g (+16)
            bfr[j].q4[0] = *(const uint4*)&Bs[cur][base];
            bfr[j].q4[1] = *(const uint4*)&Bs[cur][base + 16];
        }
#pragma unroll
        for (int i = 0; i < 4; ++i) {
            FragBF af;
            int arow = wm * 64 + i * 16 + l16;
            int base = arow * 32 + half * 8;           // K = half*8 + 2g (+16)
            af.q4[0] = *(const uint4*)&As[cur][base];
            af.q4[1] = *(const uint4*)&As[cur][base + 16];
#pragma unroll
            for (int j = 0; j < 2; ++j)
                acc[i][j] = wmma_bf16(af, bfr[j], acc[i][j]);
        }
    }

    // ---- epilogue: C layout lane(0-15)->row v, lane(16-31)->row v+8 ----
#pragma unroll
    for (int i = 0; i < 4; ++i)
#pragma unroll
        for (int j = 0; j < 2; ++j)
#pragma unroll
            for (int v = 0; v < 8; ++v) {
                int row = m0 + wm * 64 + i * 16 + v + half * 8;
                int col = n0 + wn * 32 + j * 16 + l16;
                C[(size_t)row * N + col] = acc[i][j][v];
            }
}

// ---------------------------------------------------------------------------
// Fused RMSNorm (+ optional RoPE), fp32 -> bf16. One 256-thread block per row.
// raw layout: [(b*L+l)*H + h][256]
// ---------------------------------------------------------------------------
__global__ __launch_bounds__(256) void normrope_kernel(
    const float* __restrict__ raw, const float* __restrict__ wnorm,
    u16* __restrict__ out, int L, int H, int do_rope)
{
    __shared__ float xs[256];
    __shared__ float red[8];

    const int t   = threadIdx.x;
    const int bid = blockIdx.x;
    const int h   = bid % H;
    const int l   = (bid / H) % L;
    const int b   = bid / (H * L);
    const size_t base = ((size_t)(b * L + l) * H + h) * 256;

    float x  = raw[base + t];
    float ss = x * x;
#pragma unroll
    for (int m = 16; m >= 1; m >>= 1) ss += __shfl_xor(ss, m, 32);
    if ((t & 31) == 0) red[t >> 5] = ss;
    __syncthreads();
    float tot = 0.f;
#pragma unroll
    for (int i = 0; i < 8; ++i) tot += red[i];

    float rinv = rsqrtf(tot * (1.0f / 256.0f) + 1e-6f);
    float xn   = x * rinv * wnorm[t];
    xs[t] = xn;
    __syncthreads();

    float o = xn;
    if (do_rope) {
        int   i   = t & 127;
        // inv_freq = theta^(-2i/256);  ln(1e6) = 13.815511
        float inv = __expf(-(2.0f * (float)i / 256.0f) * 13.815511f);
        float f   = (float)l * inv;
        float s, c;
        __sincosf(f, &s, &c);
        if (t < 128) o = xn * c - xs[t + 128] * s;
        else         o = xs[t - 128] * s + xn * c;
    }
    out[base + t] = f2bf(o);
}

// ---------------------------------------------------------------------------
// V transpose: vraw[(b*L+l)*256 + hd] (fp32) -> vt[(b*256+hd)*L + l] (bf16)
// ---------------------------------------------------------------------------
__global__ __launch_bounds__(256) void vtrans_kernel(
    const float* __restrict__ vraw, u16* __restrict__ vt, int L)
{
    const int t = threadIdx.x;
    const int l = blockIdx.x % L;
    const int b = blockIdx.x / L;
    float x = vraw[((size_t)(b * L) + l) * 256 + t];
    vt[((size_t)(b * 256) + t) * L + l] = f2bf(x);
}

// ---------------------------------------------------------------------------
// Flash attention. 128 threads = 4 waves; wave owns 16 query rows.
// q: [(b*L+l)*2048 + h*256 + hd] bf16   k: [(b*L+j)*256 + hd] bf16
// vt: [(b*256+hd)*L + j] bf16           ctx: same layout as q (bf16)
// ---------------------------------------------------------------------------
__global__ __launch_bounds__(128) void attn_kernel(
    const u16* __restrict__ qbf, const u16* __restrict__ kbf,
    const u16* __restrict__ vt,  u16* __restrict__ ctx, int L)
{
    __shared__ __align__(16) u16 Qs[64 * 256];
    __shared__ __align__(16) u16 Ps[4][16 * 32];

    const int tid  = threadIdx.x;
    const int lane = tid & 31;
    const int w    = tid >> 5;
    const int half = lane >> 4;
    const int l16  = lane & 15;

    const int nqb = L / 64;
    const int qb  = blockIdx.x % nqb;
    const int h   = (blockIdx.x / nqb) % 8;
    const int b   = blockIdx.x / (nqb * 8);

    // ---- stage the 64x256 Q tile into LDS ----
    const u16* qsrc = qbf + ((size_t)(b * L) + qb * 64) * 2048 + h * 256;
    for (int i = tid; i < 2048; i += 128) {          // 2048 uint4 = 64*256 bf16
        int row = (i * 8) >> 8;
        int col = (i * 8) & 255;
        *(uint4*)&Qs[row * 256 + col] = *(const uint4*)&qsrc[(size_t)row * 2048 + col];
    }
    __syncthreads();

    const int  q0 = qb * 64 + w * 16;
    const u16* kB = kbf + (size_t)b * L * 256;
    const u16* vB = vt  + (size_t)b * 256 * (size_t)L;

    v8f acc[16];
#pragma unroll
    for (int i = 0; i < 16; ++i) acc[i] = v8f_zero();
    float mrow[8], lrow[8];
#pragma unroll
    for (int v = 0; v < 8; ++v) { mrow[v] = -3.0e38f; lrow[v] = 0.f; }

    for (int ks = 0; ks < q0 + 16; ks += 32) {
        // ---- S = Q @ K^T for 32 keys (two 16x16 tiles) ----
        v8f s0 = v8f_zero(), s1 = v8f_zero();
#pragma unroll
        for (int kc = 0; kc < 8; ++kc) {
            FragBF qf;
            int qbase = (w * 16 + l16) * 256 + kc * 32 + half * 8;
            qf.q4[0] = *(const uint4*)&Qs[qbase];
            qf.q4[1] = *(const uint4*)&Qs[qbase + 16];

            FragBF kf0, kf1;
            const u16* kr0 = kB + (size_t)(ks + l16) * 256 + kc * 32 + half * 8;
            kf0.q4[0] = *(const uint4*)kr0;
            kf0.q4[1] = *(const uint4*)(kr0 + 16);
            const u16* kr1 = kr0 + 16 * 256;
            kf1.q4[0] = *(const uint4*)kr1;
            kf1.q4[1] = *(const uint4*)(kr1 + 16);

            s0 = wmma_bf16(qf, kf0, s0);
            s1 = wmma_bf16(qf, kf1, s1);
        }

        // ---- scale + causal mask + online softmax (width-16 reductions) ----
#pragma unroll
        for (int v = 0; v < 8; ++v) {
            int   row  = q0 + v + half * 8;
            int   key0 = ks + l16;
            float a = s0[v] * 0.0625f;           // 1/sqrt(256)
            float c = s1[v] * 0.0625f;
            if (key0 > row)       a = -3.0e38f;
            if (key0 + 16 > row)  c = -3.0e38f;

            float mx = fmaxf(a, c);
#pragma unroll
            for (int m = 8; m >= 1; m >>= 1) mx = fmaxf(mx, __shfl_xor(mx, m, 16));
            float mnew  = fmaxf(mrow[v], mx);
            float alpha = __expf(mrow[v] - mnew);
            float p0 = __expf(a - mnew);
            float p1 = __expf(c - mnew);
            float rs = p0 + p1;
#pragma unroll
            for (int m = 8; m >= 1; m >>= 1) rs += __shfl_xor(rs, m, 16);
            lrow[v] = lrow[v] * alpha + rs;
            mrow[v] = mnew;
#pragma unroll
            for (int nt = 0; nt < 16; ++nt) acc[nt][v] *= alpha;

            Ps[w][(v + half * 8) * 32 + l16]      = f2bf(p0);
            Ps[w][(v + half * 8) * 32 + l16 + 16] = f2bf(p1);
        }
        // Cross-lane LDS RAW inside one wave: wait for DS writes to land.
        asm volatile("s_wait_dscnt 0" ::: "memory");

        // ---- O += P @ V ----
        FragBF pf;
        int pbase = l16 * 32 + half * 8;
        pf.q4[0] = *(const uint4*)&Ps[w][pbase];
        pf.q4[1] = *(const uint4*)&Ps[w][pbase + 16];
#pragma unroll
        for (int nt = 0; nt < 16; ++nt) {
            FragBF vf;
            const u16* vr = vB + (size_t)(nt * 16 + l16) * L + ks + half * 8;
            vf.q4[0] = *(const uint4*)vr;
            vf.q4[1] = *(const uint4*)(vr + 16);
            acc[nt] = wmma_bf16(pf, vf, acc[nt]);
        }
    }

    // ---- normalize and store ctx (bf16) ----
#pragma unroll
    for (int nt = 0; nt < 16; ++nt)
#pragma unroll
        for (int v = 0; v < 8; ++v) {
            int row = q0 + v + half * 8;
            int col = nt * 16 + l16;
            float o = acc[nt][v] / lrow[v];
            ctx[((size_t)(b * L) + row) * 2048 + h * 256 + col] = f2bf(o);
        }
}

// ---------------------------------------------------------------------------
// Host side
// ---------------------------------------------------------------------------
extern "C" void kernel_launch(void* const* d_in, const int* in_sizes, int n_in,
                              void* d_out, int out_size, void* d_ws, size_t ws_size,
                              hipStream_t stream)
{
    (void)in_sizes; (void)n_in; (void)out_size; (void)ws_size;
    const int B = 2, L = 2048, D = 2048, H = 8, HD = 256;
    const int M = B * L;                 // 4096

    const float* x      = (const float*)d_in[0];
    const float* q_proj = (const float*)d_in[1];
    const float* k_proj = (const float*)d_in[2];
    const float* v_proj = (const float*)d_in[3];
    const float* o_proj = (const float*)d_in[4];
    const float* q_norm = (const float*)d_in[5];
    const float* k_norm = (const float*)d_in[6];
    float* out = (float*)d_out;

    // workspace carve-up
    char* ws = (char*)d_ws;
    size_t off = 0;
    float* qraw = (float*)(ws + off); off += (size_t)M * (H * HD) * 4;  // 33.5 MB
    float* kraw = (float*)(ws + off); off += (size_t)M * HD * 4;        //  4  MB
    float* vraw = (float*)(ws + off); off += (size_t)M * HD * 4;        //  4  MB
    u16*   qbf  = (u16*)  (ws + off); off += (size_t)M * (H * HD) * 2;  // 16.8 MB
    u16*   kbf  = (u16*)  (ws + off); off += (size_t)M * HD * 2;        //  2  MB
    u16*   vtb  = (u16*)  (ws + off); off += (size_t)B * HD * L * 2;    //  2  MB
    u16*   xbf  = (u16*)  (ws + off); off += (size_t)M * D * 2;         // 16.8 MB
    u16*   qwb  = (u16*)  (ws + off); off += (size_t)(H * HD) * D * 2;  //  8.4 MB
    u16*   kwb  = (u16*)  (ws + off); off += (size_t)HD * D * 2;        //  1  MB
    u16*   vwb  = (u16*)  (ws + off); off += (size_t)HD * D * 2;        //  1  MB
    u16*   owb  = (u16*)  (ws + off); off += (size_t)D * (H * HD) * 2;  //  8.4 MB
    // ctx aliases qraw (qraw is dead after normrope on q)
    u16*   ctx  = (u16*)qraw;

    dim3 blk256(256), blk128(128);

    // 0: one-shot fp32 -> bf16 conversions (activations + weights)
    cvt_bf16_kernel<<<dim3((M * D) / 2048), blk256, 0, stream>>>(x, xbf, M * D);
    cvt_bf16_kernel<<<dim3((H * HD * D) / 2048), blk256, 0, stream>>>(q_proj, qwb, H * HD * D);
    cvt_bf16_kernel<<<dim3((HD * D) / 2048), blk256, 0, stream>>>(k_proj, kwb, HD * D);
    cvt_bf16_kernel<<<dim3((HD * D) / 2048), blk256, 0, stream>>>(v_proj, vwb, HD * D);
    cvt_bf16_kernel<<<dim3((D * H * HD) / 2048), blk256, 0, stream>>>(o_proj, owb, D * H * HD);

    // 1-3: QKV projections (TDM-staged bf16 GEMM)
    gemm_bf16_kernel<<<dim3(H * HD / 128, M / 128), blk256, 0, stream>>>(
        xbf, qwb, qraw, M, H * HD, D);
    gemm_bf16_kernel<<<dim3(HD / 128, M / 128), blk256, 0, stream>>>(
        xbf, kwb, kraw, M, HD, D);
    gemm_bf16_kernel<<<dim3(HD / 128, M / 128), blk256, 0, stream>>>(
        xbf, vwb, vraw, M, HD, D);

    // 4-5: RMSNorm + RoPE -> bf16
    normrope_kernel<<<dim3(B * L * H), blk256, 0, stream>>>(qraw, q_norm, qbf, L, H, 1);
    normrope_kernel<<<dim3(B * L),     blk256, 0, stream>>>(kraw, k_norm, kbf, L, 1, 1);

    // 6: V -> bf16 transposed [B][HD][L]
    vtrans_kernel<<<dim3(B * L), blk256, 0, stream>>>(vraw, vtb, L);

    // 7: flash attention -> ctx bf16
    attn_kernel<<<dim3(B * H * (L / 64)), blk128, 0, stream>>>(qbf, kbf, vtb, ctx, L);

    // 8: output projection -> fp32 d_out
    gemm_bf16_kernel<<<dim3(D / 128, M / 128), blk256, 0, stream>>>(
        ctx, owb, out, M, D, H * HD);
}
